// PaliGemmaWithExpertModel_18957985644668
// MI455X (gfx1250) — matrix operand order repaired and checked by reading, not compiled, not run on hardware
//
#include <hip/hip_runtime.h>
#include <hip/hip_bf16.h>

#define B_    4
#define SV    768
#define SE    64
#define SS    832
#define WV    2048
#define WE    1024
#define NH    8
#define DH    256
#define HD    2048      // NH*DH
#define MLPV  16384
#define MLPE  4096
#define SCALE_ 0.0625f  // 256^-0.5

typedef __attribute__((ext_vector_type(16))) __bf16 v16bf;
typedef __attribute__((ext_vector_type(2)))  __bf16 v2bf;
typedef __attribute__((ext_vector_type(8)))  float  v8f;

#if defined(__has_builtin)
#  if __has_builtin(__builtin_amdgcn_cvt_pk_bf16_f32)
#    define USE_CVT_PK_BF16 1
#  endif
#endif

// pack two floats into packed bf16 (lo = a, hi = b)
__device__ __forceinline__ unsigned pack2(float a, float b) {
#ifdef USE_CVT_PK_BF16
  union { v2bf v; unsigned u; } c;
  c.v = __builtin_amdgcn_cvt_pk_bf16_f32(a, b);       // single v_cvt_pk_bf16_f32
  return c.u;
#else
  // round-half-up + byte-permute high halves: 3 VALU (2x add, 1x v_perm_b32)
  const unsigned ua = __float_as_uint(a) + 0x8000u;
  const unsigned ub = __float_as_uint(b) + 0x8000u;
  return __builtin_amdgcn_perm(ub, ua, 0x07060302u);  // {ub[31:16], ua[31:16]}
#endif
}

struct GemmCfg {
  long long aOuter, aInner; int aDiv, lda;
  long long bOuter, bInner; int bDiv, ldb;
  long long cOuter, cInner; int cDiv, ldc;
  long long rOuter, rInner; int rDiv, ldr;
  long long gOuter;         int gDiv;
  int M, N, K;
};

// ---------------------------------------------------------------------------
// Generic bf16-WMMA GEMM: C = A(MxK) * B(KxN) [+ R] [* gate(col)]
// Block tile 128x128, BK=32, 256 threads (8 waves, 4x2), wave tile 32x64.
// 8 v_wmma_f32_16x16x32_bf16 per k-step per wave; software-pipelined
// global->register prefetch overlaps VMEM latency with the matrix work.
// LDS tiles stored in WMMA fragment order (ISA 7.12.2):
//   A 16x32 tile: slot = (m%16)+16*hi, hi=(k>>3)&1, j = (k<16 ? k&7 : 8+(k&7))
//   B 32x16 tile: slot = (n%16)+16*hi, hi=(k>>3)&1, j = (k<16 ? k&7 : 8+(k&7))
// ---------------------------------------------------------------------------
__global__ __launch_bounds__(256) void gemm_bf16_wmma(
    const float* __restrict__ A, const float* __restrict__ Bm,
    float* __restrict__ C, const float* __restrict__ R,
    const float* __restrict__ G, GemmCfg cfg)
{
  const int z = blockIdx.z;
  const long long aBase = (long long)(z / cfg.aDiv) * cfg.aOuter + (long long)(z % cfg.aDiv) * cfg.aInner;
  const long long bBase = (long long)(z / cfg.bDiv) * cfg.bOuter + (long long)(z % cfg.bDiv) * cfg.bInner;
  const long long cBase = (long long)(z / cfg.cDiv) * cfg.cOuter + (long long)(z % cfg.cDiv) * cfg.cInner;
  const long long rBase = (long long)(z / cfg.rDiv) * cfg.rOuter + (long long)(z % cfg.rDiv) * cfg.rInner;
  const long long gBase = (long long)(z / cfg.gDiv) * cfg.gOuter;

  const int m0 = blockIdx.y * 128;
  const int n0 = blockIdx.x * 128;
  const int tid  = threadIdx.x;
  const int lane = tid & 31;
  const int wave = tid >> 5;
  const int wm = wave & 3, wn = wave >> 2;

  __shared__ __align__(16) unsigned short As[8 * 512];  // 128x32 bf16
  __shared__ __align__(16) unsigned short Bs[8 * 512];  // 32x128 bf16

  v8f acc[2][4];
  #pragma unroll
  for (int i = 0; i < 2; ++i)
    #pragma unroll
    for (int j = 0; j < 4; ++j)
      #pragma unroll
      for (int e = 0; e < 8; ++e) acc[i][j][e] = 0.f;

  // A staging map: thread -> (row = tid>>1, 16 k starting at (tid&1)*16)
  const int arow = tid >> 1;             // 0..127
  const int akb  = (tid & 1) << 4;       // 0 or 16
  const int ajb  = akb >> 1;             // j base: 0 or 8
  const int gmA  = m0 + arow;
  const bool okA = gmA < cfg.M;
  const int gmc  = okA ? gmA : (cfg.M - 1);
  const float zfA = okA ? 1.f : 0.f;
  unsigned short* aDst0 = &As[(arow >> 4) * 512 + ((arow & 15)     ) * 16 + ajb];
  unsigned short* aDst1 = &As[(arow >> 4) * 512 + ((arow & 15) + 16) * 16 + ajb];

  // B staging map: thread -> (col = tid&127, k 8h..8h+7 and 16+8h..16+8h+7, h = tid>>7)
  const int bn   = tid & 127;
  const int bkh  = tid >> 7;             // 0 or 1 (k base 0 / 8)
  const int gnB  = n0 + bn;
  const bool okB = gnB < cfg.N;
  const int gnc  = okB ? gnB : (cfg.N - 1);
  const float zfB = okB ? 1.f : 0.f;
  unsigned short* bDst = &Bs[(bn >> 4) * 512 + ((bn & 15) + 16 * bkh) * 16];

  // register prefetch buffers
  float4 pa0, pa1, pa2, pa3;
  float  pb[16];

  auto prefetch = [&](int k0) {
    const float* ap = A + aBase + (long long)gmc * cfg.lda + (k0 + akb);
    pa0 = *(const float4*)(ap + 0);
    pa1 = *(const float4*)(ap + 4);
    pa2 = *(const float4*)(ap + 8);
    pa3 = *(const float4*)(ap + 12);
    const float* bp = Bm + bBase + (long long)(k0 + bkh * 8) * cfg.ldb + gnc;
    #pragma unroll
    for (int j = 0; j < 16; ++j) pb[j] = bp[(j + (j & 8)) * cfg.ldb];
  };

  prefetch(0);
  const int nK = cfg.K >> 5;
  for (int kt = 0; kt < nK; ++kt) {
    __syncthreads();   // previous iteration's fragment reads complete
    {   // ---- convert + store staged A tile (2x ds_store_b128)
      uint4 r0, r1;
      r0.x = pack2(pa0.x * zfA, pa0.y * zfA); r0.y = pack2(pa0.z * zfA, pa0.w * zfA);
      r0.z = pack2(pa1.x * zfA, pa1.y * zfA); r0.w = pack2(pa1.z * zfA, pa1.w * zfA);
      r1.x = pack2(pa2.x * zfA, pa2.y * zfA); r1.y = pack2(pa2.z * zfA, pa2.w * zfA);
      r1.z = pack2(pa3.x * zfA, pa3.y * zfA); r1.w = pack2(pa3.z * zfA, pa3.w * zfA);
      *(uint4*)aDst0 = r0;
      *(uint4*)aDst1 = r1;
    }
    {   // ---- convert + store staged B tile (2x ds_store_b128)
      uint4 r0, r1;
      r0.x = pack2(pb[0]  * zfB, pb[1]  * zfB); r0.y = pack2(pb[2]  * zfB, pb[3]  * zfB);
      r0.z = pack2(pb[4]  * zfB, pb[5]  * zfB); r0.w = pack2(pb[6]  * zfB, pb[7]  * zfB);
      r1.x = pack2(pb[8]  * zfB, pb[9]  * zfB); r1.y = pack2(pb[10] * zfB, pb[11] * zfB);
      r1.z = pack2(pb[12] * zfB, pb[13] * zfB); r1.w = pack2(pb[14] * zfB, pb[15] * zfB);
      *(uint4*)bDst       = r0;
      *(uint4*)(bDst + 8) = r1;
    }
    __syncthreads();   // stores visible

    if (kt + 1 < nK) prefetch((kt + 1) << 5);   // overlap next VMEM with WMMA

    v16bf afrag[2], bfrag[4];
    #pragma unroll
    for (int mi = 0; mi < 2; ++mi) {
      const uint4* p = (const uint4*)&As[((wm << 1) + mi) * 512 + lane * 16];
      union { uint4 q[2]; v16bf v; } u;
      u.q[0] = p[0]; u.q[1] = p[1];
      afrag[mi] = u.v;
    }
    #pragma unroll
    for (int ni = 0; ni < 4; ++ni) {
      const uint4* p = (const uint4*)&Bs[((wn << 2) + ni) * 512 + lane * 16];
      union { uint4 q[2]; v16bf v; } u;
      u.q[0] = p[0]; u.q[1] = p[1];
      bfrag[ni] = u.v;
    }
    #pragma unroll
    for (int mi = 0; mi < 2; ++mi)
      #pragma unroll
      for (int ni = 0; ni < 4; ++ni)
        acc[mi][ni] = __builtin_amdgcn_wmma_f32_16x16x32_bf16(
            false, afrag[mi], false, bfrag[ni], (short)0, acc[mi][ni], false, false);
  }

  // Epilogue: D layout = lane n=(lane&15), rows m = r + 8*(lane>>4)
  #pragma unroll
  for (int mi = 0; mi < 2; ++mi) {
    const int mB = m0 + (wm << 5) + (mi << 4) + ((lane >> 4) << 3);
    #pragma unroll
    for (int ni = 0; ni < 4; ++ni) {
      const int n = n0 + (wn << 6) + (ni << 4) + (lane & 15);
      if (n < cfg.N) {
        const float gmul = G ? G[gBase + n] : 1.f;
        #pragma unroll
        for (int r = 0; r < 8; ++r) {
          const int m = mB + r;
          if (m < cfg.M) {
            float val = acc[mi][ni][r] * gmul;
            if (R) val += R[rBase + (long long)m * cfg.ldr + n];
            C[cBase + (long long)m * cfg.ldc + n] = val;
          }
        }
      }
    }
  }
}

// ---------------------------------------------------------------------------
// mod = cond(B,K) @ dw(K,N) + db(N)
__global__ __launch_bounds__(256) void mod_kernel(
    const float* __restrict__ cond, const float* __restrict__ dw,
    const float* __restrict__ db, float* __restrict__ outp, int K_, int N_)
{
  const int i = blockIdx.x * 256 + threadIdx.x;
  if (i >= B_ * N_) return;
  const int b = i / N_, n = i % N_;
  float s = db[n];
  for (int k = 0; k < K_; ++k) s += cond[b * K_ + k] * dw[(long long)k * N_ + n];
  outp[i] = s;
}

// out = rms(x) * (1 + wt[col])
__global__ __launch_bounds__(256) void rms_scale_kernel(
    const float* __restrict__ x, const float* __restrict__ wt,
    float* __restrict__ outp, int n)
{
  const long long row = blockIdx.x;
  const float* xr = x + row * n;
  float s = 0.f;
  for (int c = threadIdx.x; c < n; c += 256) { float v = xr[c]; s += v * v; }
  __shared__ float red[8];
  for (int o = 16; o >= 1; o >>= 1) s += __shfl_xor(s, o, 32);
  if ((threadIdx.x & 31) == 0) red[threadIdx.x >> 5] = s;
  __syncthreads();
  if (threadIdx.x == 0) {
    float t = 0.f; for (int i = 0; i < 8; ++i) t += red[i];
    red[0] = rsqrtf(t / n + 1e-6f);
  }
  __syncthreads();
  const float r = red[0];
  for (int c = threadIdx.x; c < n; c += 256)
    outp[row * n + c] = xr[c] * r * (1.f + wt[c]);
}

// out = rms(x) * (1 + mod[b,0W+col]) + mod[b,1W+col]   (adaLN)
__global__ __launch_bounds__(256) void rms_mod_kernel(
    const float* __restrict__ x, const float* __restrict__ mod,
    float* __restrict__ outp, int n, int rowsPerB)
{
  const long long row = blockIdx.x;
  const int b = (int)(row / rowsPerB);
  const float* xr = x + row * n;
  float s = 0.f;
  for (int c = threadIdx.x; c < n; c += 256) { float v = xr[c]; s += v * v; }
  __shared__ float red[8];
  for (int o = 16; o >= 1; o >>= 1) s += __shfl_xor(s, o, 32);
  if ((threadIdx.x & 31) == 0) red[threadIdx.x >> 5] = s;
  __syncthreads();
  if (threadIdx.x == 0) {
    float t = 0.f; for (int i = 0; i < 8; ++i) t += red[i];
    red[0] = rsqrtf(t / n + 1e-6f);
  }
  __syncthreads();
  const float r = red[0];
  const float* mb = mod + (long long)b * 3 * n;
  for (int c = threadIdx.x; c < n; c += 256)
    outp[row * n + c] = xr[c] * r * (1.f + mb[c]) + mb[n + c];
}

// RoPE on Q, pack to (B,H,S,DH)
__global__ __launch_bounds__(256) void pack_q_kernel(
    const float* __restrict__ Qv, const float* __restrict__ Qe,
    const int* __restrict__ pos, float* __restrict__ Qb)
{
  const long long i = (long long)blockIdx.x * 256 + threadIdx.x;
  if (i >= (long long)B_ * NH * SS * DH) return;
  const int d = (int)(i & 255);
  long long t = i >> 8;
  const int s = (int)(t % SS); t /= SS;
  const int h = (int)(t & 7);
  const int b = (int)(t >> 3);
  const int dp = (d < 128) ? d + 128 : d - 128;
  float x, xp;
  if (s < SV) {
    const long long rb = ((long long)b * SV + s) * HD;
    x = Qv[rb + h * DH + d]; xp = Qv[rb + h * DH + dp];
  } else {
    const long long rb = ((long long)b * SE + (s - SV)) * HD;
    x = Qe[rb + h * DH + d]; xp = Qe[rb + h * DH + dp];
  }
  const float p  = (float)pos[b * SS + s];
  const float th = p * __expf(-9.2103403719761836f * (float)(d & 127) * (1.f / 128.f));
  const float rot = (d < 128) ? -xp : xp;
  Qb[i] = x * __cosf(th) + rot * __sinf(th);
}

// RoPE on K -> Kt(B,DH,S); V -> Vb(B,S,DH)
__global__ __launch_bounds__(256) void pack_kv_kernel(
    const float* __restrict__ Kv, const float* __restrict__ Ke,
    const float* __restrict__ Vv, const float* __restrict__ Ve,
    const int* __restrict__ pos, float* __restrict__ Kt, float* __restrict__ Vb)
{
  const long long i = (long long)blockIdx.x * 256 + threadIdx.x;
  if (i >= (long long)B_ * SS * DH) return;
  const int d = (int)(i & 255);
  long long t = i >> 8;
  const int s = (int)(t % SS);
  const int b = (int)(t / SS);
  const int dp = (d < 128) ? d + 128 : d - 128;
  float kx, kp, vx;
  if (s < SV) {
    const long long r = ((long long)b * SV + s) * DH;
    kx = Kv[r + d]; kp = Kv[r + dp]; vx = Vv[r + d];
  } else {
    const long long r = ((long long)b * SE + (s - SV)) * DH;
    kx = Ke[r + d]; kp = Ke[r + dp]; vx = Ve[r + d];
  }
  const float p  = (float)pos[b * SS + s];
  const float th = p * __expf(-9.2103403719761836f * (float)(d & 127) * (1.f / 128.f));
  const float rot = (d < 128) ? -kp : kp;
  Kt[((long long)b * DH + d) * SS + s] = kx * __cosf(th) + rot * __sinf(th);
  Vb[((long long)b * SS + s) * DH + d] = vx;
}

// wave-per-row softmax of scale*logit + mask, in place. 832 = 26*32
__global__ __launch_bounds__(256) void softmax_kernel(
    float* __restrict__ logits, const float* __restrict__ mask)
{
  const int wave = threadIdx.x >> 5, lane = threadIdx.x & 31;
  const long long row = (long long)blockIdx.x * 8 + wave;  // [0, B*NH*SS)
  const int q = (int)(row % SS);
  const int b = (int)(row / ((long long)NH * SS));
  float* rp = logits + row * SS;
  const float* mp = mask + ((long long)b * SS + q) * SS;
  float vals[26];
  float mx = -3.4e38f;
  #pragma unroll
  for (int i = 0; i < 26; ++i) {
    const float v = rp[lane + 32 * i] * SCALE_ + mp[lane + 32 * i];
    vals[i] = v; mx = fmaxf(mx, v);
  }
  for (int o = 16; o >= 1; o >>= 1) mx = fmaxf(mx, __shfl_xor(mx, o, 32));
  float sum = 0.f;
  #pragma unroll
  for (int i = 0; i < 26; ++i) { vals[i] = __expf(vals[i] - mx); sum += vals[i]; }
  for (int o = 16; o >= 1; o >>= 1) sum += __shfl_xor(sum, o, 32);
  const float inv = 1.f / sum;
  #pragma unroll
  for (int i = 0; i < 26; ++i) rp[lane + 32 * i] = vals[i] * inv;
}

// g = gelu_tanh(g) * u
__global__ __launch_bounds__(256) void geglu_kernel(
    float* __restrict__ g, const float* __restrict__ u, long long n)
{
  const long long i = (long long)blockIdx.x * 256 + threadIdx.x;
  if (i >= n) return;
  const float x = g[i];
  const float t = tanhf(0.7978845608f * (x + 0.044715f * x * x * x));
  g[i] = 0.5f * x * (1.f + t) * u[i];
}

// ---------------------------------------------------------------------------
static inline GemmCfg mkcfg(int M, int N, int K, int lda, int ldb, int ldc) {
  GemmCfg c{};
  c.M = M; c.N = N; c.K = K; c.lda = lda; c.ldb = ldb; c.ldc = ldc; c.ldr = ldc;
  c.aDiv = c.bDiv = c.cDiv = c.rDiv = c.gDiv = 1;
  return c;
}

extern "C" void kernel_launch(void* const* d_in, const int* in_sizes, int n_in,
                              void* d_out, int out_size, void* d_ws, size_t ws_size,
                              hipStream_t stream) {
  (void)in_sizes; (void)n_in; (void)out_size; (void)ws_size;
  const float* embeds_vlm = (const float*)d_in[0];
  const float* embeds_exp = (const float*)d_in[1];
  const float* cond       = (const float*)d_in[2];
  const float* vlm_ln1_w  = (const float*)d_in[3];
  const float* vlm_ln2_w  = (const float*)d_in[4];
  const float* vlm_q_w    = (const float*)d_in[5];
  const float* vlm_k_w    = (const float*)d_in[6];
  const float* vlm_v_w    = (const float*)d_in[7];
  const float* vlm_o_w    = (const float*)d_in[8];
  const float* vlm_gate_w = (const float*)d_in[9];
  const float* vlm_up_w   = (const float*)d_in[10];
  const float* vlm_down_w = (const float*)d_in[11];
  const float* exp_ln1_dw = (const float*)d_in[12];
  const float* exp_ln1_db = (const float*)d_in[13];
  const float* exp_ln2_dw = (const float*)d_in[14];
  const float* exp_ln2_db = (const float*)d_in[15];
  const float* exp_q_w    = (const float*)d_in[16];
  const float* exp_k_w    = (const float*)d_in[17];
  const float* exp_v_w    = (const float*)d_in[18];
  const float* exp_o_w    = (const float*)d_in[19];
  const float* exp_gate_w = (const float*)d_in[20];
  const float* exp_up_w   = (const float*)d_in[21];
  const float* exp_down_w = (const float*)d_in[22];
  const int*   pos_ids    = (const int*)d_in[23];
  const float* attn_mask  = (const float*)d_in[24];
  float* out = (float*)d_out;

  float* w = (float*)d_ws;
  size_t off = 0;
  auto alloc = [&](size_t n) { float* p = w + off; off += n; return p; };
  float* hv     = alloc((size_t)B_ * SV * WV);
  float* he     = alloc((size_t)B_ * SE * WE);
  float* mod1   = alloc((size_t)B_ * 3 * WE);
  float* mod2   = alloc((size_t)B_ * 3 * WE);
  float* Qv     = alloc((size_t)B_ * SV * HD);
  float* Qe     = alloc((size_t)B_ * SE * HD);
  float* Kv     = alloc((size_t)B_ * SV * DH);
  float* Ke     = alloc((size_t)B_ * SE * DH);
  float* Vv     = alloc((size_t)B_ * SV * DH);
  float* Ve     = alloc((size_t)B_ * SE * DH);
  float* Qb     = alloc((size_t)B_ * NH * SS * DH);
  float* Kt     = alloc((size_t)B_ * DH * SS);
  float* Vb     = alloc((size_t)B_ * SS * DH);
  float* logits = alloc((size_t)B_ * NH * SS * SS);
  float* attn   = alloc((size_t)B_ * SS * HD);
  float* hv1    = alloc((size_t)B_ * SV * WV);
  float* he1    = alloc((size_t)B_ * SE * WE);
  float* nv     = alloc((size_t)B_ * SV * WV);
  float* ne     = alloc((size_t)B_ * SE * WE);
  float* gv     = alloc((size_t)B_ * SV * MLPV);
  float* uv     = alloc((size_t)B_ * SV * MLPV);
  float* ge     = alloc((size_t)B_ * SE * MLPE);
  float* ue     = alloc((size_t)B_ * SE * MLPE);

  auto gemm = [&](const float* A, const float* Bm, float* C, const float* R,
                  const float* G, GemmCfg cfg, int Z) {
    dim3 grid((cfg.N + 127) / 128, (cfg.M + 127) / 128, Z);
    gemm_bf16_wmma<<<grid, dim3(256), 0, stream>>>(A, Bm, C, R, G, cfg);
  };

  // --- modulation vectors
  {
    const int tot = B_ * 3 * WE;
    mod_kernel<<<(tot + 255) / 256, 256, 0, stream>>>(cond, exp_ln1_dw, exp_ln1_db, mod1, WE, 3 * WE);
    mod_kernel<<<(tot + 255) / 256, 256, 0, stream>>>(cond, exp_ln2_dw, exp_ln2_db, mod2, WE, 3 * WE);
  }
  // --- pre-attention norms
  rms_scale_kernel<<<B_ * SV, 256, 0, stream>>>(embeds_vlm, vlm_ln1_w, hv, WV);
  rms_mod_kernel<<<B_ * SE, 256, 0, stream>>>(embeds_exp, mod1, he, WE, SE);

  // --- QKV projections
  gemm(hv, vlm_q_w, Qv, nullptr, nullptr, mkcfg(B_ * SV, HD, WV, WV, HD, HD), 1);
  gemm(hv, vlm_k_w, Kv, nullptr, nullptr, mkcfg(B_ * SV, DH, WV, WV, DH, DH), 1);
  gemm(hv, vlm_v_w, Vv, nullptr, nullptr, mkcfg(B_ * SV, DH, WV, WV, DH, DH), 1);
  gemm(he, exp_q_w, Qe, nullptr, nullptr, mkcfg(B_ * SE, HD, WE, WE, HD, HD), 1);
  gemm(he, exp_k_w, Ke, nullptr, nullptr, mkcfg(B_ * SE, DH, WE, WE, DH, DH), 1);
  gemm(he, exp_v_w, Ve, nullptr, nullptr, mkcfg(B_ * SE, DH, WE, WE, DH, DH), 1);

  // --- RoPE + pack
  {
    const long long tq = (long long)B_ * NH * SS * DH;
    pack_q_kernel<<<(unsigned)((tq + 255) / 256), 256, 0, stream>>>(Qv, Qe, pos_ids, Qb);
    const long long tk = (long long)B_ * SS * DH;
    pack_kv_kernel<<<(unsigned)((tk + 255) / 256), 256, 0, stream>>>(Kv, Ke, Vv, Ve, pos_ids, Kt, Vb);
  }

  // --- logits = Q @ K^T  (per b,h)
  {
    GemmCfg c = mkcfg(SS, SS, DH, DH, SS, SS);
    c.aOuter = (long long)SS * DH;
    c.bOuter = (long long)DH * SS; c.bDiv = NH;
    c.cOuter = (long long)SS * SS;
    gemm(Qb, Kt, logits, nullptr, nullptr, c, B_ * NH);
  }
  softmax_kernel<<<B_ * NH * SS / 8, 256, 0, stream>>>(logits, attn_mask);

  // --- attn = probs @ V  -> (B,S,H*DH)
  {
    GemmCfg c = mkcfg(SS, DH, SS, SS, DH, HD);
    c.aOuter = (long long)SS * SS;
    c.bOuter = (long long)SS * DH; c.bDiv = NH;
    c.cOuter = (long long)SS * HD; c.cInner = DH; c.cDiv = NH;
    gemm(logits, Vb, attn, nullptr, nullptr, c, B_ * NH);
  }

  // --- output projections + residual (+ g1 gate for expert)
  {
    GemmCfg c = mkcfg(SV, WV, HD, HD, WV, WV);
    c.aOuter = (long long)SS * HD;
    c.cOuter = (long long)SV * WV;
    c.rOuter = (long long)SV * WV;
    gemm(attn, vlm_o_w, hv1, embeds_vlm, nullptr, c, B_);
  }
  {
    GemmCfg c = mkcfg(SE, WE, HD, HD, WE, WE);
    c.aOuter = (long long)SS * HD;
    c.cOuter = (long long)SE * WE;
    c.rOuter = (long long)SE * WE;
    c.gOuter = 3 * WE;
    gemm(attn + (long long)SV * HD, exp_o_w, he1, embeds_exp, mod1 + 2 * WE, c, B_);
  }

  // --- post-attention norms
  rms_scale_kernel<<<B_ * SV, 256, 0, stream>>>(hv1, vlm_ln2_w, nv, WV);
  rms_mod_kernel<<<B_ * SE, 256, 0, stream>>>(he1, mod2, ne, WE, SE);

  // --- VLM MLP
  gemm(nv, vlm_gate_w, gv, nullptr, nullptr, mkcfg(B_ * SV, MLPV, WV, WV, MLPV, MLPV), 1);
  gemm(nv, vlm_up_w,   uv, nullptr, nullptr, mkcfg(B_ * SV, MLPV, WV, WV, MLPV, MLPV), 1);
  {
    const long long n = (long long)B_ * SV * MLPV;
    geglu_kernel<<<(unsigned)((n + 255) / 256), 256, 0, stream>>>(gv, uv, n);
  }
  gemm(gv, vlm_down_w, out, hv1, nullptr, mkcfg(B_ * SV, WV, MLPV, MLPV, WV, WV), 1);

  // --- Expert MLP (with g2 gate)
  gemm(ne, exp_gate_w, ge, nullptr, nullptr, mkcfg(B_ * SE, MLPE, WE, WE, MLPE, MLPE), 1);
  gemm(ne, exp_up_w,   ue, nullptr, nullptr, mkcfg(B_ * SE, MLPE, WE, WE, MLPE, MLPE), 1);
  {
    const long long n = (long long)B_ * SE * MLPE;
    geglu_kernel<<<(unsigned)((n + 255) / 256), 256, 0, stream>>>(ge, ue, n);
  }
  {
    GemmCfg c = mkcfg(SE, WE, MLPE, MLPE, WE, WE);
    c.aOuter = (long long)SE * MLPE;
    c.cOuter = (long long)SE * WE;
    c.rOuter = (long long)SE * WE;
    c.gOuter = 3 * WE;
    gemm(ge, exp_down_w, out + (size_t)B_ * SV * WV, he1, mod2 + 2 * WE, c, B_);
  }
}
